// DecisionTransformer_52432960749652
// MI455X (gfx1250) — matrix operand order repaired and checked by reading, not compiled, not run on hardware
//
#include <hip/hip_runtime.h>
#include <cstdint>
#include <cstddef>

typedef __bf16 bf16_t;
typedef __attribute__((ext_vector_type(16))) __bf16 v16bf;
typedef __attribute__((ext_vector_type(8)))  __bf16 v8bf;
typedef __attribute__((ext_vector_type(8)))  float  v8f;
typedef __attribute__((ext_vector_type(4)))  unsigned int u32x4;
typedef __attribute__((ext_vector_type(8)))  int i32x8;
typedef __attribute__((ext_vector_type(4)))  int i32x4;

union FragBF { v16bf v; v8bf h[2]; };

static __device__ __forceinline__ v8f vzero8f() {
  v8f z = {0.f,0.f,0.f,0.f,0.f,0.f,0.f,0.f};
  return z;
}

// ---- CDNA5 TDM support (this toolchain: 6-arg builtin; fallback to sync) ----
#if defined(__has_builtin)
#  if __has_builtin(__builtin_amdgcn_tensor_load_to_lds)
#    define HAVE_TDM 1
#  endif
#endif
#ifndef HAVE_TDM
#  define HAVE_TDM 0
#endif

#if defined(__has_builtin)
#  if __has_builtin(__builtin_amdgcn_s_wait_tensorcnt)
#    define WAIT_TENSORCNT(n) __builtin_amdgcn_s_wait_tensorcnt((short)(n))
#  endif
#endif
#ifndef WAIT_TENSORCNT
#  define WAIT_TENSORCNT(n) asm volatile("s_wait_tensorcnt %0" :: "i"(n) : "memory")
#endif

#if HAVE_TDM
// 2D tile DMA: global (row-major, stride in elements of 2B) -> LDS, with
// optional row padding so LDS rows keep a padded stride for conflict-free
// b128 fragment loads. D# layout per CDNA5 ISA §8.3/8.4.
//  pad_interval_code: pad after 2^(c+1) DWORDs; pad_amount_code: (c+1) DWORDs.
static __device__ __forceinline__ void tdm_load_2d_bf16(
    unsigned lds_off, const void* gptr, int tile_w, int tile_h,
    unsigned long long stride_elems, unsigned pi_code, unsigned pa_code)
{
  const unsigned long long ga = (unsigned long long)(uintptr_t)gptr;
  u32x4 g0;
  g0[0] = 1u;                                            // count=1, user mode
  g0[1] = lds_off;                                       // lds_addr
  g0[2] = (unsigned)(ga & 0xFFFFFFFFu);                  // global_addr[31:0]
  g0[3] = (unsigned)((ga >> 32) & 0x01FFFFFFu) | (2u << 30); // addr[56:32], type=2

  const unsigned td0 = 0x40000000u;                      // huge tensor dims:
  const unsigned td1 = 0x40000000u;                      // tiles always in range
  i32x8 g1;
  g1[0] = (int)((1u << 16)                               // data_size = 2 bytes
              | (1u << 20)                               // pad_enable
              | (pi_code << 22) | (pa_code << 25));
  g1[1] = (int)((td0 & 0xFFFFu) << 16);                  // barrier_addr=0 | td0.lo
  g1[2] = (int)(((td0 >> 16) & 0xFFFFu) | ((td1 & 0xFFFFu) << 16));
  g1[3] = (int)(((td1 >> 16) & 0xFFFFu) | (((unsigned)tile_w & 0xFFFFu) << 16));
  g1[4] = (int)((unsigned)tile_h & 0xFFFFu);             // tile_dim1 | tile_dim2=0
  g1[5] = (int)(unsigned)(stride_elems & 0xFFFFFFFFull); // dim0_stride[31:0]
  g1[6] = (int)(unsigned)((stride_elems >> 32) & 0xFFFFull); // dim0_stride[47:32]
  g1[7] = 0;
  const i32x4 z4 = {0, 0, 0, 0};
  const i32x8 z8 = {0, 0, 0, 0, 0, 0, 0, 0};
  __builtin_amdgcn_tensor_load_to_lds(g0, g1, z4, z4, z8, 0);
}
#endif

// problem constants
constexpr int kB  = 8;
constexpr int kS  = 512;
constexpr int kD  = 1024;
constexpr int kH  = 16;
constexpr int kHD = 64;          // head dim
constexpr int kT  = 3 * kS;      // 1536 tokens per batch
constexpr int kM  = kB * kT;     // 12288 rows
constexpr int kD4 = 4 * kD;      // 4096
constexpr int kL  = 6;
constexpr int kAct = 32;

// ---------------------------------------------------------------------------
// fp32 -> bf16 cast (weights)
// ---------------------------------------------------------------------------
__global__ __launch_bounds__(256) void k_cast_bf16(const float* __restrict__ in,
                                                   bf16_t* __restrict__ out, size_t n) {
  size_t i = (size_t)blockIdx.x * blockDim.x + threadIdx.x;
  size_t stride = (size_t)gridDim.x * blockDim.x;
  for (; i < n; i += stride) out[i] = (bf16_t)in[i];
}

// ---------------------------------------------------------------------------
// Embedding: rtg/state/action projections + time embedding, interleave 3 tokens
// ---------------------------------------------------------------------------
__global__ __launch_bounds__(256) void k_embed(
    const float* __restrict__ states, const float* __restrict__ actions,
    const float* __restrict__ rtg, const int* __restrict__ tsteps,
    const float* __restrict__ et,  const float* __restrict__ erw,
    const float* __restrict__ erb, const float* __restrict__ esw,
    const float* __restrict__ esb, const float* __restrict__ eaw,
    const float* __restrict__ eab, float* __restrict__ x)
{
  const int bs = blockIdx.x;            // 0 .. B*S-1
  const int b  = bs / kS, s = bs % kS;
  const int t  = tsteps[bs];
  const float r0 = rtg[bs];
  const size_t rowbase = ((size_t)b * kT + 3 * s) * kD;
  for (int d = threadIdx.x; d < kD; d += 256) {
    const float te = et[(size_t)t * kD + d];
    float re = r0 * erw[d] + erb[d] + te;                 // K = 1
    float st = esb[d] + te;                               // K = 128
    #pragma unroll 4
    for (int k = 0; k < 128; ++k) st += states[(size_t)bs * 128 + k] * esw[(size_t)k * kD + d];
    float ac = eab[d] + te;                               // K = 32
    #pragma unroll 4
    for (int k = 0; k < 32; ++k) ac += actions[(size_t)bs * 32 + k] * eaw[(size_t)k * kD + d];
    x[rowbase + d]          = re;
    x[rowbase + kD + d]     = st;
    x[rowbase + 2 * kD + d] = ac;
  }
}

// ---------------------------------------------------------------------------
// LayerNorm over D=1024; optional fp32 output (may alias input) and bf16 output
// ---------------------------------------------------------------------------
__global__ __launch_bounds__(256) void k_ln(const float* __restrict__ xin,
    const float* __restrict__ g, const float* __restrict__ bta,
    float* of, bf16_t* ob)
{
  __shared__ float red[256];
  const int row = blockIdx.x;
  const int tid = threadIdx.x;
  const float* xr = xin + (size_t)row * kD;
  float s = 0.f;
  for (int i = tid; i < kD; i += 256) s += xr[i];
  red[tid] = s; __syncthreads();
  for (int st = 128; st > 0; st >>= 1) { if (tid < st) red[tid] += red[tid + st]; __syncthreads(); }
  const float mu = red[0] * (1.f / kD);
  __syncthreads();
  float vv = 0.f;
  for (int i = tid; i < kD; i += 256) { float d0 = xr[i] - mu; vv += d0 * d0; }
  red[tid] = vv; __syncthreads();
  for (int st = 128; st > 0; st >>= 1) { if (tid < st) red[tid] += red[tid + st]; __syncthreads(); }
  const float rstd = rsqrtf(red[0] * (1.f / kD) + 1e-5f);
  for (int i = tid; i < kD; i += 256) {
    float y = (xr[i] - mu) * rstd * g[i] + bta[i];
    if (of) of[(size_t)row * kD + i] = y;
    if (ob) ob[(size_t)row * kD + i] = (bf16_t)y;
  }
}

// ---------------------------------------------------------------------------
// bf16 WMMA GEMM:  out = A[MxK] @ W[KxN] + bias, epilogue:
//   epi==0: store bf16   epi==1: exact GELU then store bf16   epi==2: resid += val
// Tiles: 128x128x32, 256 threads (8 waves), wave tile 32x64 (2x4 WMMA frags).
// A tile staged by the Tensor Data Mover, double-buffered (TENSORcnt overlap).
// ---------------------------------------------------------------------------
constexpr int GBM = 128, GBN = 128, GBK = 32;
constexpr int LDA = 40;   // padded bf16 stride: 64B row + 16B pad (TDM pad 3/3)
constexpr int LDB = 40;

__global__ __launch_bounds__(256) void k_gemm(
    const bf16_t* __restrict__ A, const bf16_t* __restrict__ W,
    const float* __restrict__ bias, bf16_t* __restrict__ outb,
    float* __restrict__ resid, int M, int N, int K, int epi)
{
  __shared__ __align__(16) bf16_t sA2[2][GBM * LDA];
  __shared__ __align__(16) bf16_t sB[GBN * LDB];

  const int tid  = threadIdx.x;
  const int wid  = tid >> 5;
  const int lane = tid & 31;
  const int hl   = lane >> 4;      // 16-lane half select
  const int l16  = lane & 15;
  const int bm = blockIdx.x * GBM;
  const int bn = blockIdx.y * GBN;
  const int waveM = (wid >> 1) * 32;   // 0,32,64,96
  const int waveN = (wid & 1) * 64;    // 0,64

  v8f acc[2][4];
  #pragma unroll
  for (int mi = 0; mi < 2; ++mi)
    #pragma unroll
    for (int ni = 0; ni < 4; ++ni) acc[mi][ni] = vzero8f();

#if HAVE_TDM
  if (wid == 0) {   // prologue: DMA first A tile (128 rows x 32 cols)
    tdm_load_2d_bf16((unsigned)(uintptr_t)&sA2[0][0],
                     A + (size_t)bm * K, GBK, GBM,
                     (unsigned long long)K, /*16 DW*/3, /*4 DW*/3);
  }
#endif

  int p = 0;
  for (int k0 = 0; k0 < K; k0 += GBK, p ^= 1) {
    __syncthreads();   // previous iteration done with sB and the other A buffer
    { // stage B tile 32x128, transposed into n-major sB[n][k]
      const int kk = tid >> 3, nc = (tid & 7) * 16;
      const bf16_t* gW = W + (size_t)(k0 + kk) * N + bn + nc;
      v8bf b0 = *(const v8bf*)gW;
      v8bf b1 = *(const v8bf*)(gW + 8);
      #pragma unroll
      for (int i = 0; i < 8; ++i) sB[(nc + i) * LDB + kk]     = b0[i];
      #pragma unroll
      for (int i = 0; i < 8; ++i) sB[(nc + 8 + i) * LDB + kk] = b1[i];
      if (k0 + GBK < K) __builtin_prefetch(gW + (size_t)GBK * N, 0, 0);
    }
#if HAVE_TDM
    if (wid == 0) {
      if (k0 + GBK < K) {   // prefetch next A tile into the other buffer
        tdm_load_2d_bf16((unsigned)(uintptr_t)&sA2[p ^ 1][0],
                         A + (size_t)bm * K + (k0 + GBK), GBK, GBM,
                         (unsigned long long)K, 3, 3);
        WAIT_TENSORCNT(1);  // oldest (current buffer) complete
      } else {
        WAIT_TENSORCNT(0);
      }
    }
#else
    { // fallback: cooperative synchronous A staging
      const int r = tid >> 1, c = (tid & 1) * 16;
      const bf16_t* gA = A + (size_t)(bm + r) * K + k0 + c;
      *(v8bf*)&sA2[p][r * LDA + c]     = *(const v8bf*)gA;
      *(v8bf*)&sA2[p][r * LDA + c + 8] = *(const v8bf*)(gA + 8);
    }
#endif
    __syncthreads();   // A tile (TDM) + B tile visible to all waves

    FragBF afr[2];
    #pragma unroll
    for (int mi = 0; mi < 2; ++mi) {
      const int r = waveM + mi * 16 + l16;
      afr[mi].h[0] = *(const v8bf*)&sA2[p][r * LDA + hl * 8];
      afr[mi].h[1] = *(const v8bf*)&sA2[p][r * LDA + hl * 8 + 16];
    }
    FragBF bfr[4];
    #pragma unroll
    for (int ni = 0; ni < 4; ++ni) {
      const int c = waveN + ni * 16 + l16;
      bfr[ni].h[0] = *(const v8bf*)&sB[c * LDB + hl * 8];
      bfr[ni].h[1] = *(const v8bf*)&sB[c * LDB + hl * 8 + 16];
    }
    #pragma unroll
    for (int mi = 0; mi < 2; ++mi)
      #pragma unroll
      for (int ni = 0; ni < 4; ++ni)
        acc[mi][ni] = __builtin_amdgcn_wmma_f32_16x16x32_bf16(
            false, afr[mi].v, false, bfr[ni].v, (short)0, acc[mi][ni], false, false);
  }

  // epilogue
  #pragma unroll
  for (int mi = 0; mi < 2; ++mi) {
    #pragma unroll
    for (int ni = 0; ni < 4; ++ni) {
      const int col = bn + waveN + ni * 16 + l16;
      const float bs = bias ? bias[col] : 0.f;
      #pragma unroll
      for (int v = 0; v < 8; ++v) {
        const int row = bm + waveM + mi * 16 + v + hl * 8;
        float val = acc[mi][ni][v] + bs;
        if (epi == 2) {
          resid[(size_t)row * N + col] += val;
        } else {
          if (epi == 1) val = 0.5f * val * (1.0f + erff(val * 0.70710678118654752f));
          outb[(size_t)row * N + col] = (bf16_t)val;
        }
      }
    }
  }
}

// ---------------------------------------------------------------------------
// Flash attention (causal), bf16 WMMA for QK^T and P*V, fp32 online softmax.
// Grid: (T/64, B*H). Block: 128 threads (4 waves); wave owns 16 query rows.
// K block staged by TDM; V block transposed cooperatively.
// ---------------------------------------------------------------------------
constexpr int ATQ = 64, ATK = 32;
constexpr int LDK = 72;   // 128B row + 16B pad (TDM pad codes 4/3)

__global__ __launch_bounds__(128) void k_attn(
    const bf16_t* __restrict__ Qp, const bf16_t* __restrict__ Kp,
    const bf16_t* __restrict__ Vp, bf16_t* __restrict__ Yp)
{
  __shared__ __align__(16) bf16_t sK[ATK * LDK];    // [key][d]
  __shared__ __align__(16) bf16_t sVt[kHD][40];     // [d][key] (transposed)
  __shared__ __align__(16) bf16_t sP[4][16][40];    // per-wave P tile [q][key]

  const int bh = blockIdx.y;
  const int b = bh / kH, h = bh % kH;
  const int qb = blockIdx.x * ATQ;
  const int tid = threadIdx.x, wid = tid >> 5, lane = tid & 31;
  const int hl = lane >> 4, l16 = lane & 15;
  const int qrow = qb + wid * 16;

  const bf16_t* qbase = Qp + (size_t)b * kT * kD + (size_t)h * kHD;
  const bf16_t* kbase = Kp + (size_t)b * kT * kD + (size_t)h * kHD;
  const bf16_t* vbase = Vp + (size_t)b * kT * kD + (size_t)h * kHD;

  // load Q fragments (k-dim = head dim 64 -> two 16x32 A fragments)
  FragBF qa[2];
  #pragma unroll
  for (int kk = 0; kk < 2; ++kk) {
    const bf16_t* pq = qbase + (size_t)(qrow + l16) * kD + kk * 32 + hl * 8;
    qa[kk].h[0] = *(const v8bf*)pq;
    qa[kk].h[1] = *(const v8bf*)(pq + 16);
  }

  v8f o[4];
  #pragma unroll
  for (int ni = 0; ni < 4; ++ni) o[ni] = vzero8f();
  float m8[8], l8[8];
  #pragma unroll
  for (int v = 0; v < 8; ++v) { m8[v] = -1e30f; l8[v] = 0.f; }

  const int nkb = (qb + ATQ) / ATK;    // cover keys up to qb+63; mask the rest
  for (int kb = 0; kb < nkb; ++kb) {
    const int kb0 = kb * ATK;
    __syncthreads();
    { // stage V block transposed -> sVt[d][key]
      const int r = tid >> 2, c = (tid & 3) * 16;
      const bf16_t* gv = vbase + (size_t)(kb0 + r) * kD + c;
      v8bf v0 = *(const v8bf*)gv;
      v8bf v1 = *(const v8bf*)(gv + 8);
      #pragma unroll
      for (int i = 0; i < 8; ++i) sVt[c + i][r]     = v0[i];
      #pragma unroll
      for (int i = 0; i < 8; ++i) sVt[c + 8 + i][r] = v1[i];
    }
#if HAVE_TDM
    if (wid == 0) {   // DMA K block 32x64 (row-major, padded rows)
      tdm_load_2d_bf16((unsigned)(uintptr_t)&sK[0],
                       kbase + (size_t)kb0 * kD, kHD, ATK,
                       (unsigned long long)kD, /*32 DW*/4, /*4 DW*/3);
      WAIT_TENSORCNT(0);
    }
#else
    { // fallback: cooperative K staging
      const int r = tid >> 2, c = (tid & 3) * 16;
      const bf16_t* gk = kbase + (size_t)(kb0 + r) * kD + c;
      *(v8bf*)&sK[r * LDK + c]     = *(const v8bf*)gk;
      *(v8bf*)&sK[r * LDK + c + 8] = *(const v8bf*)(gk + 8);
    }
#endif
    __syncthreads();

    // S = Q K^T : two 16x16 score fragments over this 32-key block
    v8f sfr[2] = { vzero8f(), vzero8f() };
    #pragma unroll
    for (int nf = 0; nf < 2; ++nf) {
      #pragma unroll
      for (int kk = 0; kk < 2; ++kk) {
        FragBF kf;
        const int c = nf * 16 + l16;               // key within block
        kf.h[0] = *(const v8bf*)&sK[c * LDK + kk * 32 + hl * 8];
        kf.h[1] = *(const v8bf*)&sK[c * LDK + kk * 32 + hl * 8 + 16];
        sfr[nf] = __builtin_amdgcn_wmma_f32_16x16x32_bf16(
            false, qa[kk].v, false, kf.v, (short)0, sfr[nf], false, false);
      }
    }

    // scale + causal mask
    #pragma unroll
    for (int nf = 0; nf < 2; ++nf) {
      #pragma unroll
      for (int v = 0; v < 8; ++v) {
        const int row = qrow + hl * 8 + v;
        const int col = kb0 + nf * 16 + l16;
        float sv = sfr[nf][v] * 0.125f;            // 1/sqrt(64)
        if (col > row) sv = -__builtin_inff();
        sfr[nf][v] = sv;
      }
    }

    // online softmax (rows live in 16-lane groups)
    float alpha[8];
    #pragma unroll
    for (int v = 0; v < 8; ++v) {
      float mx = fmaxf(sfr[0][v], sfr[1][v]);
      #pragma unroll
      for (int off = 1; off < 16; off <<= 1) mx = fmaxf(mx, __shfl_xor(mx, off, 32));
      const float mn = fmaxf(m8[v], mx);
      alpha[v] = __expf(m8[v] - mn);
      m8[v] = mn;
    }
    #pragma unroll
    for (int nf = 0; nf < 2; ++nf)
      #pragma unroll
      for (int v = 0; v < 8; ++v) sfr[nf][v] = __expf(sfr[nf][v] - m8[v]);
    #pragma unroll
    for (int v = 0; v < 8; ++v) {
      float rs = sfr[0][v] + sfr[1][v];
      #pragma unroll
      for (int off = 1; off < 16; off <<= 1) rs += __shfl_xor(rs, off, 32);
      l8[v] = l8[v] * alpha[v] + rs;
    }
    #pragma unroll
    for (int ni = 0; ni < 4; ++ni)
      #pragma unroll
      for (int v = 0; v < 8; ++v) o[ni][v] *= alpha[v];

    // redistribute P (C-layout) into A-layout via per-wave LDS tile
    #pragma unroll
    for (int nf = 0; nf < 2; ++nf)
      #pragma unroll
      for (int v = 0; v < 8; ++v)
        sP[wid][hl * 8 + v][nf * 16 + l16] = (bf16_t)sfr[nf][v];
    asm volatile("s_wait_dscnt 0" ::: "memory");   // per-wave LDS RAW ordering

    FragBF pa;
    pa.h[0] = *(const v8bf*)&sP[wid][l16][hl * 8];
    pa.h[1] = *(const v8bf*)&sP[wid][l16][hl * 8 + 16];

    // O += P @ V  (k-dim = 32 keys, 4 N-fragments over head dim)
    #pragma unroll
    for (int nf = 0; nf < 4; ++nf) {
      FragBF vf;
      const int c = nf * 16 + l16;                 // d index
      vf.h[0] = *(const v8bf*)&sVt[c][hl * 8];
      vf.h[1] = *(const v8bf*)&sVt[c][hl * 8 + 16];
      o[nf] = __builtin_amdgcn_wmma_f32_16x16x32_bf16(
          false, pa.v, false, vf.v, (short)0, o[nf], false, false);
    }
  }

  // normalize + store bf16
  bf16_t* ybase = Yp + (size_t)b * kT * kD + (size_t)h * kHD;
  #pragma unroll
  for (int nf = 0; nf < 4; ++nf) {
    #pragma unroll
    for (int v = 0; v < 8; ++v) {
      const int row = qrow + hl * 8 + v;
      const int col = nf * 16 + l16;
      ybase[(size_t)row * kD + col] = (bf16_t)(o[nf][v] / l8[v]);
    }
  }
}

// ---------------------------------------------------------------------------
// Head: state tokens (x[:,1::3,:]) @ pred_w + pred_b -> tanh. One wave/row.
// ---------------------------------------------------------------------------
__global__ __launch_bounds__(32) void k_head(const float* __restrict__ x,
    const float* __restrict__ pw, const float* __restrict__ pb,
    float* __restrict__ out)
{
  const int bs = blockIdx.x;             // 0 .. B*S-1
  const int b = bs / kS, s = bs % kS;
  const float* row = x + ((size_t)b * kT + 3 * s + 1) * kD;
  const int a = threadIdx.x;             // 0..31
  float acc = pb[a];
  for (int k = 0; k < kD; ++k) acc += row[k] * pw[(size_t)k * kAct + a];
  out[(size_t)bs * kAct + a] = tanhf(acc);
}

// ---------------------------------------------------------------------------
// Orchestration
// ---------------------------------------------------------------------------
extern "C" void kernel_launch(void* const* d_in, const int* in_sizes, int n_in,
                              void* d_out, int out_size, void* d_ws, size_t ws_size,
                              hipStream_t stream) {
  (void)in_sizes; (void)n_in; (void)out_size; (void)ws_size;
  const float* states  = (const float*)d_in[0];
  const float* actions = (const float*)d_in[1];
  const float* rtg     = (const float*)d_in[2];
  const int*   tsteps  = (const int*)d_in[3];
  int idx = 4;
  const float* embed_timestep = (const float*)d_in[idx++];
  const float* embed_return_w = (const float*)d_in[idx++];
  const float* embed_return_b = (const float*)d_in[idx++];
  const float* embed_state_w  = (const float*)d_in[idx++];
  const float* embed_state_b  = (const float*)d_in[idx++];
  const float* embed_action_w = (const float*)d_in[idx++];
  const float* embed_action_b = (const float*)d_in[idx++];
  const float* embed_ln_g     = (const float*)d_in[idx++];
  const float* embed_ln_b     = (const float*)d_in[idx++];
  const float* blk[kL][16];
  for (int l = 0; l < kL; ++l)
    for (int j = 0; j < 16; ++j) blk[l][j] = (const float*)d_in[idx++];
  const float* pred_w = (const float*)d_in[idx++];
  const float* pred_b = (const float*)d_in[idx++];

  // workspace carve-up
  char* wsp = (char*)d_ws;
  auto take = [&](size_t bytes) -> void* {
    void* p = (void*)wsp; wsp += (bytes + 255) & ~(size_t)255; return p;
  };
  float*  x    = (float*) take((size_t)kM * kD * sizeof(float));
  bf16_t* lnb  = (bf16_t*)take((size_t)kM * kD * sizeof(bf16_t));
  bf16_t* qbuf = (bf16_t*)take((size_t)kM * kD * sizeof(bf16_t));
  bf16_t* kbuf = (bf16_t*)take((size_t)kM * kD * sizeof(bf16_t));
  bf16_t* vbuf = (bf16_t*)take((size_t)kM * kD * sizeof(bf16_t));
  bf16_t* ybuf = (bf16_t*)take((size_t)kM * kD * sizeof(bf16_t));
  bf16_t* hbuf = (bf16_t*)take((size_t)kM * kD4 * sizeof(bf16_t));
  bf16_t* wslot= (bf16_t*)take((size_t)kD * kD4 * sizeof(bf16_t));

  const dim3 gGemmD (kM / GBM, kD  / GBN);
  const dim3 gGemmD4(kM / GBM, kD4 / GBN);
  const dim3 gAttn(kT / ATQ, kB * kH);

  // 1) embeddings + embed layernorm (in place on fp32 residual stream)
  k_embed<<<kB * kS, 256, 0, stream>>>(states, actions, rtg, tsteps,
      embed_timestep, embed_return_w, embed_return_b, embed_state_w,
      embed_state_b, embed_action_w, embed_action_b, x);
  k_ln<<<kM, 256, 0, stream>>>(x, embed_ln_g, embed_ln_b, x, nullptr);

  // 2) transformer blocks
  for (int l = 0; l < kL; ++l) {
    const float *ln1g = blk[l][0], *ln1b = blk[l][1];
    const float *wq = blk[l][2], *bq = blk[l][3];
    const float *wk = blk[l][4], *bk = blk[l][5];
    const float *wv = blk[l][6], *bv = blk[l][7];
    const float *wo = blk[l][8], *bo = blk[l][9];
    const float *ln2g = blk[l][10], *ln2b = blk[l][11];
    const float *w1 = blk[l][12], *b1 = blk[l][13];
    const float *w2 = blk[l][14], *b2 = blk[l][15];

    k_ln<<<kM, 256, 0, stream>>>(x, ln1g, ln1b, nullptr, lnb);

    k_cast_bf16<<<1024, 256, 0, stream>>>(wq, wslot, (size_t)kD * kD);
    k_gemm<<<gGemmD, 256, 0, stream>>>(lnb, wslot, bq, qbuf, nullptr, kM, kD, kD, 0);
    k_cast_bf16<<<1024, 256, 0, stream>>>(wk, wslot, (size_t)kD * kD);
    k_gemm<<<gGemmD, 256, 0, stream>>>(lnb, wslot, bk, kbuf, nullptr, kM, kD, kD, 0);
    k_cast_bf16<<<1024, 256, 0, stream>>>(wv, wslot, (size_t)kD * kD);
    k_gemm<<<gGemmD, 256, 0, stream>>>(lnb, wslot, bv, vbuf, nullptr, kM, kD, kD, 0);

    k_attn<<<gAttn, 128, 0, stream>>>(qbuf, kbuf, vbuf, ybuf);

    k_cast_bf16<<<1024, 256, 0, stream>>>(wo, wslot, (size_t)kD * kD);
    k_gemm<<<gGemmD, 256, 0, stream>>>(ybuf, wslot, bo, nullptr, x, kM, kD, kD, 2);

    k_ln<<<kM, 256, 0, stream>>>(x, ln2g, ln2b, nullptr, lnb);

    k_cast_bf16<<<2048, 256, 0, stream>>>(w1, wslot, (size_t)kD * kD4);
    k_gemm<<<gGemmD4, 256, 0, stream>>>(lnb, wslot, b1, hbuf, nullptr, kM, kD4, kD, 1);
    k_cast_bf16<<<2048, 256, 0, stream>>>(w2, wslot, (size_t)kD4 * kD);
    k_gemm<<<gGemmD, 256, 0, stream>>>(hbuf, wslot, b2, nullptr, x, kM, kD, kD4, 2);
  }

  // 3) prediction head over state tokens
  k_head<<<kB * kS, 32, 0, stream>>>(x, pred_w, pred_b, (float*)d_out);
}
